// RFAConv_8899172237565
// MI455X (gfx1250) — matrix operand order repaired and checked by reading, not compiled, not run on hardware
//
#include <hip/hip_runtime.h>
#include <hip/hip_bf16.h>

// ---------------------------------------------------------------------------
// RFAConv fused pipeline for gfx1250 (MI455X).
//   Phase 0: weight fp32->bf16 + BN2/bias folding
//   Phase 1: fused avgpool/softmax-attn/depthwise-conv/BN1/ReLU producer -> V (bf16)
//   Phase 2: WMMA bf16 GEMM  out[oc,pos] = W[oc,kk] * V[pos,kk]^T
//            M=256, K=1152, N=73728; block tile 256Mx128N so V is read ONCE.
//            Staging via Tensor Data Mover (tensor_load_to_lds) double-buffered.
// ---------------------------------------------------------------------------

typedef __attribute__((ext_vector_type(16))) __bf16       v16bf;
typedef __attribute__((ext_vector_type(8)))  float        v8f;
typedef __attribute__((ext_vector_type(4)))  unsigned int u32x4;

union FragBF {
    v16bf v;
    u32x4 q[2];
};

#define CB   128            // input channels
#define OCB  256            // output channels
#define HB   96
#define WB   96
#define BB   8
#define K2   9
#define KK   (CB * K2)      // 1152 reduction length
#define NPOS (BB * HB * WB) // 73728 spatial positions

#define LDSROW 40           // padded LDS row stride (bf16 elems): 32 data + 8 pad
#define ATILE_ELEMS (256 * LDSROW)
#define BTILE_ELEMS (128 * LDSROW)
#define SMEM_ELEMS  (2 * ATILE_ELEMS + 2 * BTILE_ELEMS)
#define SMEM_BYTES  (SMEM_ELEMS * 2)

#if defined(__has_builtin)
#if __has_builtin(__builtin_amdgcn_tensor_load_to_lds) && __has_builtin(__builtin_amdgcn_s_wait_tensorcnt)
#define RFA_TDM 1
#endif
#endif
#ifndef RFA_TDM
#define RFA_TDM 0
#endif

// ---------------------------------------------------------------------------
// Phase 0: convert conv_w to bf16 (layout [oc][kk] == flat OIHW) and fold
// conv bias + BN2 into per-oc alpha/beta:  y = relu(acc*alpha + beta)
// ---------------------------------------------------------------------------
__global__ void rfa_prep_kernel(const float* __restrict__ conv_w,
                                const float* __restrict__ conv_b,
                                const float* __restrict__ bn2_g,
                                const float* __restrict__ bn2_b,
                                const float* __restrict__ bn2_m,
                                const float* __restrict__ bn2_v,
                                __bf16* __restrict__ Wbf,
                                float* __restrict__ alpha,
                                float* __restrict__ beta) {
    int i = blockIdx.x * blockDim.x + threadIdx.x;
    if (i < OCB * KK) {
        Wbf[i] = (__bf16)conv_w[i];
    }
    if (i < OCB) {
        float sc = bn2_g[i] * rsqrtf(bn2_v[i] + 1e-5f);
        alpha[i] = sc;
        beta[i]  = conv_b[i] * sc + bn2_b[i] - bn2_m[i] * sc;
    }
}

// ---------------------------------------------------------------------------
// Phase 1: producer.  One thread per (b,c,h,w); emits 9 bf16 values of
// V[pos][c*9 + n]  (row-major over kk, stride KK per position).
// ---------------------------------------------------------------------------
__global__ __launch_bounds__(256) void rfa_producer_kernel(
        const float* __restrict__ x,
        const float* __restrict__ w1,      // [C][9]
        const float* __restrict__ gf_w,    // [C*9][9] (flat OIHW, I=1)
        const float* __restrict__ bn1_g,
        const float* __restrict__ bn1_b,
        const float* __restrict__ bn1_m,
        const float* __restrict__ bn1_v,
        __bf16* __restrict__ V) {
    int idx = blockIdx.x * blockDim.x + threadIdx.x;
    if (idx >= BB * CB * HB * WB) return;

    int w = idx % WB;
    int t = idx / WB;
    int h = t % HB;
    t /= HB;
    int c = t % CB;
    int b = t / CB;

    const float* xp = x + ((size_t)(b * CB + c) * HB) * WB;

    // 3x3 zero-padded neighborhood (shared by avgpool and depthwise conv)
    float xn[9];
    float s = 0.0f;
#pragma unroll
    for (int dy = -1; dy <= 1; ++dy) {
#pragma unroll
        for (int dx = -1; dx <= 1; ++dx) {
            int hh = h + dy, ww = w + dx;
            float v = 0.0f;
            if (hh >= 0 && hh < HB && ww >= 0 && ww < WB) v = xp[hh * WB + ww];
            xn[(dy + 1) * 3 + (dx + 1)] = v;
            s += v;
        }
    }
    s *= (1.0f / 9.0f);   // count_include_pad avgpool

    // attention logits = pooled * w1[c, n]; softmax over n
    int base = c * K2;
    float att[9];
    float mx = -3.4e38f;
#pragma unroll
    for (int n = 0; n < 9; ++n) {
        att[n] = s * w1[base + n];
        mx = fmaxf(mx, att[n]);
    }
    float sum = 0.0f;
#pragma unroll
    for (int n = 0; n < 9; ++n) {
        att[n] = __expf(att[n] - mx);
        sum += att[n];
    }
    float inv = 1.0f / sum;

    // depthwise conv (mult 9) + BN1 + ReLU, reweighted by attention
    int pos = (b * HB + h) * WB + w;
    __bf16* vp = V + (size_t)pos * KK + base;
#pragma unroll
    for (int n = 0; n < 9; ++n) {
        const float* g = gf_w + (size_t)(base + n) * K2;
        float f = 0.0f;
#pragma unroll
        for (int j = 0; j < 9; ++j) f = fmaf(g[j], xn[j], f);
        float sc = bn1_g[base + n] * rsqrtf(bn1_v[base + n] + 1e-5f);
        f = (f - bn1_m[base + n]) * sc + bn1_b[base + n];
        f = fmaxf(f, 0.0f) * att[n] * inv;
        vp[n] = (__bf16)f;
    }
}

// ---------------------------------------------------------------------------
// TDM descriptor construction (CDNA5 ISA ch.8): 2D tile, 2-byte elements,
// row length 32 elems, tensor row stride 1152 elems, LDS padding of
// 4 dwords after every 16 dwords (reproduces the LDSROW=40 layout).
// ---------------------------------------------------------------------------
#if RFA_TDM
typedef unsigned int tdm_u32x4 __attribute__((ext_vector_type(4)));
typedef int          tdm_i32x4 __attribute__((ext_vector_type(4)));
typedef int          tdm_i32x8 __attribute__((ext_vector_type(8)));

__device__ __forceinline__ void tdm_load_2d(unsigned long long gaddr,
                                            unsigned lds_byte_addr,
                                            unsigned tensor_rows,
                                            unsigned tile_rows) {
    tdm_u32x4 g0;
    g0[0] = 1u;                                             // count=1, user D#
    g0[1] = lds_byte_addr;                                  // lds_addr
    g0[2] = (unsigned)(gaddr & 0xffffffffull);              // global_addr lo
    g0[3] = (unsigned)((gaddr >> 32) & 0x01ffffffull)       // global_addr hi (57b)
            | (2u << 30);                                   // type = 2 (image)
    tdm_i32x8 g1;
    g1[0] = (int)((1u << 16)        // data_size = 1 -> 2 bytes
                | (1u << 20)        // pad_enable
                | (3u << 22)        // pad_interval: 16 dwords
                | (3u << 25));      // pad_amount: 4 dwords
    g1[1] = (int)(((unsigned)KK & 0xffffu) << 16);          // tensor_dim0 lo16
    g1[2] = (int)(((unsigned)KK >> 16)                      // tensor_dim0 hi16
                | ((tensor_rows & 0xffffu) << 16));         // tensor_dim1 lo16
    g1[3] = (int)((tensor_rows >> 16)                       // tensor_dim1 hi16
                | (32u << 16));                             // tile_dim0 = 32
    g1[4] = (int)(tile_rows & 0xffffu);                     // tile_dim1; tile_dim2=0
    g1[5] = (int)KK;                                        // tensor_dim0_stride lo32
    g1[6] = 0;                                              // stride hi; dim1_stride=0
    g1[7] = 0;
    tdm_i32x4 z4 = {};
#if __has_include(<hip/amd_detail/amd_gfx1250_TDM.h>)
    tdm_i32x8 z8 = {};
    __builtin_amdgcn_tensor_load_to_lds(g0, g1, z4, z4, z8, 0);
#else
    __builtin_amdgcn_tensor_load_to_lds(g0, g1, z4, z4, 0);
#endif
}
#endif  // RFA_TDM

// ---------------------------------------------------------------------------
// Phase 2: WMMA bf16 GEMM.
//   Block: 256 threads (8 waves), tile 256M x 128N, waves in 4x2 grid of
//   64x64 tiles (4x4 WMMA 16x16x32 units each).  K staged in chunks of 32
//   through double-buffered LDS filled by the TDM.
// ---------------------------------------------------------------------------
__global__ __launch_bounds__(256) void rfa_gemm_kernel(
        const __bf16* __restrict__ A,   // [256][1152] bf16 weights
        const __bf16* __restrict__ Bv,  // [73728][1152] bf16 V
        const float* __restrict__ alpha,
        const float* __restrict__ beta,
        float* __restrict__ out) {      // [8][256][96][96] fp32
    extern __shared__ __attribute__((aligned(16))) __bf16 smem[];

    const int tid  = threadIdx.x;
    const int wave = tid >> 5;
    const int lane = tid & 31;
    const int waveM = wave & 3;   // 4 M-slices of 64 rows -> 256
    const int waveN = wave >> 1 >> 1;  // wave>>2: 2 N-slices of 64 cols -> 128
    const int no_base = blockIdx.x * 128;
    const int lm  = lane & 15;
    const int l16 = lane >> 4;

    v8f acc[4][4];
#pragma unroll
    for (int mi = 0; mi < 4; ++mi)
#pragma unroll
        for (int ni = 0; ni < 4; ++ni) acc[mi][ni] = {};

#if RFA_TDM
    // prologue: DMA chunk 0 into buffer 0
    if (wave == 0) {
        tdm_load_2d((unsigned long long)A, 0u, OCB, OCB);
        tdm_load_2d((unsigned long long)Bv + (size_t)no_base * KK * 2,
                    (unsigned)(2 * ATILE_ELEMS * 2), NPOS, 128);
        __builtin_amdgcn_s_wait_tensorcnt(0);
    }
    __syncthreads();
#endif

    int buf = 0;
    for (int kc = 0; kc < KK; kc += 32) {
        const __bf16* As = smem + buf * ATILE_ELEMS;
        const __bf16* Bs = smem + 2 * ATILE_ELEMS + buf * BTILE_ELEMS;

#if RFA_TDM
        // DMA next chunk into the other buffer while we compute this one
        if (wave == 0 && kc + 32 < KK) {
            unsigned nb = buf ^ 1;
            tdm_load_2d((unsigned long long)A + (size_t)(kc + 32) * 2,
                        nb * (ATILE_ELEMS * 2), OCB, OCB);
            tdm_load_2d((unsigned long long)Bv +
                            ((size_t)no_base * KK + kc + 32) * 2,
                        (unsigned)(2 * ATILE_ELEMS * 2) + nb * (BTILE_ELEMS * 2),
                        NPOS, 128);
        }
#else
        // synchronous fallback staging: 16B chunks, 256 threads
        {
            __bf16* Asw = smem + buf * ATILE_ELEMS;
            __bf16* Bsw = smem + 2 * ATILE_ELEMS + buf * BTILE_ELEMS;
#pragma unroll
            for (int it = 0; it < 4; ++it) {
                int chunk = tid + it * 256;          // 1024 chunks: 256 rows x 4
                int row = chunk >> 2, col = (chunk & 3) * 8;
                *(u32x4*)&Asw[row * LDSROW + col] =
                    *(const u32x4*)&A[(size_t)row * KK + kc + col];
            }
#pragma unroll
            for (int it = 0; it < 2; ++it) {
                int chunk = tid + it * 256;          // 512 chunks: 128 rows x 4
                int row = chunk >> 2, col = (chunk & 3) * 8;
                *(u32x4*)&Bsw[row * LDSROW + col] =
                    *(const u32x4*)&Bv[(size_t)(no_base + row) * KK + kc + col];
            }
        }
        __syncthreads();
#endif

        // A fragments: lane m = lane%16; K-half base = (lane/16)*8, +16
        FragBF af[4];
#pragma unroll
        for (int mi = 0; mi < 4; ++mi) {
            int m  = waveM * 64 + mi * 16 + lm;
            int bk = l16 * 8;
            af[mi].q[0] = *(const u32x4*)&As[m * LDSROW + bk];
            af[mi].q[1] = *(const u32x4*)&As[m * LDSROW + bk + 16];
        }
        // B fragments: lane n = lane%16; K-half base = (lane/16)*8, +16
        FragBF bfr[4];
#pragma unroll
        for (int ni = 0; ni < 4; ++ni) {
            int n  = waveN * 64 + ni * 16 + lm;
            int bk = l16 * 8;
            bfr[ni].q[0] = *(const u32x4*)&Bs[n * LDSROW + bk];
            bfr[ni].q[1] = *(const u32x4*)&Bs[n * LDSROW + bk + 16];
        }

#pragma unroll
        for (int mi = 0; mi < 4; ++mi)
#pragma unroll
            for (int ni = 0; ni < 4; ++ni)
                acc[mi][ni] = __builtin_amdgcn_wmma_f32_16x16x32_bf16(
                    false, af[mi].v, false, bfr[ni].v,
                    (short)0, acc[mi][ni], false, false);

#if RFA_TDM
        if (wave == 0) __builtin_amdgcn_s_wait_tensorcnt(0);
#endif
        __syncthreads();
        buf ^= 1;
    }

    // epilogue: y = relu(acc*alpha[oc] + beta[oc]); scatter to NCHW fp32
#pragma unroll
    for (int mi = 0; mi < 4; ++mi) {
#pragma unroll
        for (int ni = 0; ni < 4; ++ni) {
            int n_t = waveN * 64 + ni * 16 + lm;
            int pos = no_base + n_t;
            int bq  = pos / (HB * WB);
            int hw  = pos % (HB * WB);
#pragma unroll
            for (int r = 0; r < 8; ++r) {
                int oc = waveM * 64 + mi * 16 + l16 * 8 + r;
                float v = fmaf(acc[mi][ni][r], alpha[oc], beta[oc]);
                v = fmaxf(v, 0.0f);
                out[(size_t)(bq * OCB + oc) * (HB * WB) + hw] = v;
            }
        }
    }
}

// ---------------------------------------------------------------------------
extern "C" void kernel_launch(void* const* d_in, const int* in_sizes, int n_in,
                              void* d_out, int out_size, void* d_ws, size_t ws_size,
                              hipStream_t stream) {
    const float* x      = (const float*)d_in[0];
    const float* w1     = (const float*)d_in[1];
    const float* gf_w   = (const float*)d_in[2];
    const float* bn1_g  = (const float*)d_in[3];
    const float* bn1_b  = (const float*)d_in[4];
    const float* bn1_m  = (const float*)d_in[5];
    const float* bn1_v  = (const float*)d_in[6];
    const float* conv_w = (const float*)d_in[7];
    const float* conv_b = (const float*)d_in[8];
    const float* bn2_g  = (const float*)d_in[9];
    const float* bn2_b  = (const float*)d_in[10];
    const float* bn2_m  = (const float*)d_in[11];
    const float* bn2_v  = (const float*)d_in[12];
    float* out = (float*)d_out;

    // workspace layout
    char* ws = (char*)d_ws;
    const size_t wbf_bytes = (size_t)OCB * KK * sizeof(__bf16);      //   576 KB
    const size_t v_bytes   = (size_t)NPOS * KK * sizeof(__bf16);     // ~162 MB
    __bf16* Wbf   = (__bf16*)ws;
    __bf16* V     = (__bf16*)(ws + wbf_bytes);
    float*  alpha = (float*)(ws + wbf_bytes + v_bytes);
    float*  betap = alpha + OCB;

    // Phase 0: weight convert + BN fold
    {
        int total  = OCB * KK;
        int blocks = (total + 255) / 256;
        rfa_prep_kernel<<<blocks, 256, 0, stream>>>(conv_w, conv_b, bn2_g, bn2_b,
                                                    bn2_m, bn2_v, Wbf, alpha, betap);
    }
    // Phase 1: fused producer -> V
    {
        int total  = BB * CB * HB * WB;
        int blocks = (total + 255) / 256;
        rfa_producer_kernel<<<blocks, 256, 0, stream>>>(x, w1, gf_w, bn1_g, bn1_b,
                                                        bn1_m, bn1_v, V);
    }
    // Phase 2: WMMA GEMM + epilogue (V read exactly once from HBM)
    {
        int blocks = NPOS / 128;   // 576
        rfa_gemm_kernel<<<blocks, 256, SMEM_BYTES, stream>>>(Wbf, V, alpha, betap, out);
    }
}
